// MLP_30537217474768
// MI455X (gfx1250) — matrix operand order repaired and checked by reading, not compiled, not run on hardware
//
#include <hip/hip_runtime.h>
#include <hip/hip_bf16.h>

typedef __attribute__((ext_vector_type(16))) _Float16 v16h;
typedef __attribute__((ext_vector_type(8)))  _Float16 v8h;
typedef __attribute__((ext_vector_type(8)))  float    v8f;

#define NBP   (32 * 512)   // B * S1 workgroups
#define S2V   64           // rows per tile (S2)
#define KD    64           // reduction dim (SIZE3)
#define HD    64           // output cols (HIDDEN)
#define LSTR  72           // padded LDS row stride in f16 (144B: 16B-aligned, 16-lane conflict-free)

__global__ __launch_bounds__(128)
void mlp_pool_wmma(const float* __restrict__ in,            // (BP, 64, 64) f32
                   const unsigned char* __restrict__ mask,  // (BP, 64) bool
                   const float* __restrict__ W,             // (64, 64) [h][f] f32
                   const float* __restrict__ bias,          // (64,) f32
                   float* __restrict__ out)                 // (BP, 64, 128) f32
{
    __shared__ _Float16 ldsA[S2V * LSTR];   // f16 activations, row-major [v][f]
    __shared__ _Float16 ldsW[HD * LSTR];    // f16 weights, [h][f]
    __shared__ float    ldsBias[HD];
    __shared__ unsigned char ldsMask[S2V];
    __shared__ float    ldsPool[4][HD];     // per-wave column maxima
    __shared__ float    ldsPoolF[HD];       // final pool
    __shared__ int      anyFlag;

    const int tid = threadIdx.x;
    const int bp  = blockIdx.x;

    const float* __restrict__ inB = in + (size_t)bp * (S2V * KD);
    const unsigned char* __restrict__ mB = mask + (size_t)bp * S2V;
    float* __restrict__ outB = out + (size_t)bp * (S2V * 2 * HD);

    if (tid == 0) anyFlag = 0;

    // ---- Stage: f32 -> f16 into LDS (activation tile + W), bias, mask ----
    for (int i = tid; i < (S2V * KD) / 4; i += 128) {
        const int o   = i * 4;
        const int row = o >> 6;          // /64
        const int col = o & 63;
        const int d   = row * LSTR + col;
        float4 a = ((const float4*)inB)[i];
        ldsA[d + 0] = (_Float16)a.x; ldsA[d + 1] = (_Float16)a.y;
        ldsA[d + 2] = (_Float16)a.z; ldsA[d + 3] = (_Float16)a.w;
        float4 w = ((const float4*)W)[i];
        ldsW[d + 0] = (_Float16)w.x; ldsW[d + 1] = (_Float16)w.y;
        ldsW[d + 2] = (_Float16)w.z; ldsW[d + 3] = (_Float16)w.w;
    }
    if (tid < HD)  ldsBias[tid] = bias[tid];
    if (tid < S2V) {
        unsigned char m = mB[tid];
        ldsMask[tid] = m;
        if (m) anyFlag = 1;              // benign race: only ever writes 1
    }
    __syncthreads();

    // ---- GEMM: each wave owns a 16-row M-strip x all 64 cols ----
    const int wave = tid >> 5;           // 0..3 -> M strip
    const int lane = tid & 31;
    const int lm   = lane & 15;          // row (A) / col (B,C) within 16
    const int lh   = lane >> 4;          // K-half selector

    v8f acc[4] = {};                     // four 16x16 f32 accumulator tiles
    const int rowA = wave * 16 + lm;

#pragma unroll
    for (int ks = 0; ks < 2; ++ks) {     // K = 64 in two K=32 steps
        // A fragment: 16-bit A 16x32 layout (lanes 0-15: K 0-7 & 16-23; lanes 16-31: +8)
        union { v16h v; v8h h[2]; } afrag;
        const int c0 = ks * 32 + lh * 8;
        afrag.h[0] = *(const v8h*)&ldsA[rowA * LSTR + c0];
        afrag.h[1] = *(const v8h*)&ldsA[rowA * LSTR + c0 + 16];
#pragma unroll
        for (int nt = 0; nt < 4; ++nt) {
            // B fragment: 32x16, lane holds col h = nt*16+lm, K = ks*32 + lh*8 + {0..7, 16..23}
            union { v16h v; v8h h[2]; } bfrag;
            const int hcol = nt * 16 + lm;
            const int fb   = ks * 32 + lh * 8;
            bfrag.h[0] = *(const v8h*)&ldsW[hcol * LSTR + fb];
            bfrag.h[1] = *(const v8h*)&ldsW[hcol * LSTR + fb + 16];
            acc[nt] = __builtin_amdgcn_wmma_f32_16x16x32_f16(
                /*neg_a=*/false, afrag.v, /*neg_b=*/false, bfrag.v,
                /*c_mod=*/(short)0, acc[nt], /*reuse_a=*/false, /*reuse_b=*/false);
        }
    }

    // ---- Epilogue: bias + leaky relu + mask + store + per-column max ----
#pragma unroll
    for (int nt = 0; nt < 4; ++nt) {
        const int hcol = nt * 16 + lm;
        const float bh = ldsBias[hcol];
        float pm = -__builtin_inff();
#pragma unroll
        for (int r = 0; r < 8; ++r) {
            const int v = wave * 16 + lh * 8 + r;   // C/D layout: VGPR r, lane-half lh
            float x = acc[nt][r] + bh;
            x = (x > 0.0f) ? x : 0.01f * x;         // leaky_relu
            const bool mv = (ldsMask[v] != 0);
            outB[v * 128 + hcol] = mv ? x : 0.0f;   // masked "scatter" half
            pm = fmaxf(pm, mv ? x : -__builtin_inff());
        }
        const float other = __shfl_xor(pm, 16, 32); // fold lane halves (same column)
        pm = fmaxf(pm, other);
        if (lh == 0) ldsPool[wave][hcol] = pm;
    }
    __syncthreads();

    // ---- Cross-wave pool reduction + any_mask fixup ----
    if (tid < HD) {
        float m = fmaxf(fmaxf(ldsPool[0][tid], ldsPool[1][tid]),
                        fmaxf(ldsPool[2][tid], ldsPool[3][tid]));
        ldsPoolF[tid] = anyFlag ? m : 0.0f;
    }
    __syncthreads();

    // ---- Broadcast pool into second half of every row (vectorized) ----
    for (int i = tid; i < S2V * (HD / 4); i += 128) {
        const int v  = i >> 4;          // /16 float4s per row
        const int c4 = (i & 15) * 4;
        float4 pv = *(const float4*)&ldsPoolF[c4];
        *(float4*)&outB[v * 128 + HD + c4] = pv;
    }
}

extern "C" void kernel_launch(void* const* d_in, const int* in_sizes, int n_in,
                              void* d_out, int out_size, void* d_ws, size_t ws_size,
                              hipStream_t stream) {
    (void)in_sizes; (void)n_in; (void)d_ws; (void)ws_size; (void)out_size;
    const float*         in   = (const float*)d_in[0];
    const unsigned char* mask = (const unsigned char*)d_in[1];  // jax bool -> 1 byte
    const float*         W    = (const float*)d_in[2];
    const float*         bias = (const float*)d_in[3];
    float*               out  = (float*)d_out;

    dim3 grid(NBP);     // 16384 workgroups: one per (b, s1)
    dim3 block(128);    // 4 wave32
    hipLaunchKernelGGL(mlp_pool_wmma, grid, block, 0, stream, in, mask, W, bias, out);
}
